// RelMultiHeadAttention_29128468201743
// MI455X (gfx1250) — matrix operand order repaired and checked by reading, not compiled, not run on hardware
//
#include <hip/hip_runtime.h>

// ---------------------------------------------------------------------------
// Types for CDNA5 WMMA (gfx1250, wave32)
// ---------------------------------------------------------------------------
typedef __attribute__((ext_vector_type(16))) __bf16 v16bf;
typedef __attribute__((ext_vector_type(8)))  float  v8f;
typedef __attribute__((ext_vector_type(4)))  unsigned v4u;
typedef __attribute__((ext_vector_type(4)))  int    v4i;
typedef __attribute__((ext_vector_type(4)))  float  v4f;

union FragU { v16bf v; v4u q[2]; };

__device__ __forceinline__ v8f wmma_bf16(v16bf a, v16bf b, v8f c) {
  // D = A(16x32 bf16) x B(32x16 bf16) + C(16x16 f32)
  return __builtin_amdgcn_wmma_f32_16x16x32_bf16(false, a, false, b,
                                                 (short)0, c, false, false);
}

// ---------------------------------------------------------------------------
// Async global -> LDS copy (CDNA5 GLOBAL_LOAD_ASYNC_TO_LDS_B128, ASYNCcnt).
// Builtin signature on this toolchain: (as1 v4i*, as3 v4i*, imm offset, imm cpol)
// Guarded: falls back to load+ds_store if the toolchain lacks the builtins.
// ---------------------------------------------------------------------------
#if __has_builtin(__builtin_amdgcn_global_load_async_to_lds_b128) && \
    __has_builtin(__builtin_amdgcn_s_wait_asynccnt)
#define HAVE_ASYNC_LDS 1
#else
#define HAVE_ASYNC_LDS 0
#endif

#if HAVE_ASYNC_LDS
typedef __attribute__((address_space(1))) v4i* gas_v4i_p;
typedef __attribute__((address_space(3))) v4i* las_v4i_p;
__device__ __forceinline__ void async16(const __bf16* g, __bf16* l) {
  __builtin_amdgcn_global_load_async_to_lds_b128((gas_v4i_p)g, (las_v4i_p)l, 0, 0);
}
__device__ __forceinline__ void async_wait0() {
  __builtin_amdgcn_s_wait_asynccnt(0);
}
#else
__device__ __forceinline__ void async16(const __bf16* g, __bf16* l) {
  *(v4u*)l = *(const v4u*)g;
}
__device__ __forceinline__ void async_wait0() {}
#endif

// ---------------------------------------------------------------------------
// fp32 -> bf16 conversion (vectorized)
// ---------------------------------------------------------------------------
__global__ void to_bf16_vec(const float* __restrict__ in, __bf16* __restrict__ out, int n) {
  int i = (blockIdx.x * 256 + threadIdx.x) * 4;
  if (i < n) {
    v4f f = *(const v4f*)(in + i);
    out[i + 0] = (__bf16)f.x;
    out[i + 1] = (__bf16)f.y;
    out[i + 2] = (__bf16)f.z;
    out[i + 3] = (__bf16)f.w;
  }
}

// W[rows][cols] fp32 -> Wt[cols][rows] bf16  (so WMMA B-fragments read contiguously)
__global__ void transpose_to_bf16(const float* __restrict__ in, __bf16* __restrict__ out,
                                  int rows, int cols) {
  size_t i = (size_t)blockIdx.x * 256 + threadIdx.x;
  if (i < (size_t)rows * cols) {
    int r = (int)(i / cols), c = (int)(i % cols);
    out[(size_t)c * rows + r] = (__bf16)in[i];
  }
}

// ---------------------------------------------------------------------------
// Generic bf16 WMMA GEMM:  C[m][n] = scale * sum_k A[m][k]*Bt[n][k] + bias1[n]
// optional second output out2 = bf16(C + bias2[n]);  optional fp32 output outf.
// Batched over blockIdx.z = zb*NH + zh with element strides (asb,ash) etc.
// Block tile 128x128, 8 waves, each wave 32x64 (2x4 WMMA tiles), K-step 32.
// ---------------------------------------------------------------------------
__global__ __launch_bounds__(256) void gemm_bt_bf16(
    const __bf16* __restrict__ A, int lda, long long asb, long long ash,
    const __bf16* __restrict__ Bt, int ldb, long long bsb, long long bsh,
    const float* __restrict__ bias1, const float* __restrict__ bias2,
    __bf16* __restrict__ out1, __bf16* __restrict__ out2,
    float* __restrict__ outf,
    int ldc, long long csb, long long csh,
    int Kd, int NH, float scale)
{
  __shared__ __bf16 sA[128][40];   // pad 32->40: rows stay 16B aligned, no bank conflict
  __shared__ __bf16 sB[128][40];

  const int t  = threadIdx.x;
  const int zb = blockIdx.z / NH, zh = blockIdx.z % NH;
  A  += (size_t)zb * asb + (size_t)zh * ash;
  Bt += (size_t)zb * bsb + (size_t)zh * bsh;
  const size_t cofs = (size_t)zb * csb + (size_t)zh * csh;

  const int m0 = blockIdx.y * 128, n0 = blockIdx.x * 128;
  const int w = t >> 5, lane = t & 31, lt = lane & 15, hl = lane >> 4;
  const int wm = w & 3, wn = w >> 2;            // 4 waves along M, 2 along N
  const int lrow = t >> 1, lcol = (t & 1) * 16; // staging: 2 threads per row

  const v8f vzero = {0.f,0.f,0.f,0.f,0.f,0.f,0.f,0.f};
  v8f acc[2][4];
  for (int i = 0; i < 2; i++) for (int j = 0; j < 4; j++) acc[i][j] = vzero;

  for (int k0 = 0; k0 < Kd; k0 += 32) {
    const __bf16* ap = A  + (size_t)(m0 + lrow) * lda + k0 + lcol;
    const __bf16* bp = Bt + (size_t)(n0 + lrow) * ldb + k0 + lcol;
    if (k0 + 32 < Kd) {                         // global_prefetch_b8 next K-step
      __builtin_prefetch(ap + 32, 0, 3);
      __builtin_prefetch(bp + 32, 0, 3);
    }
    __syncthreads();                            // prior iteration reads complete
    // DMA the A/B tiles straight into LDS (no VGPR round trip on CDNA5)
    async16(ap,     &sA[lrow][lcol]);
    async16(ap + 8, &sA[lrow][lcol + 8]);
    async16(bp,     &sB[lrow][lcol]);
    async16(bp + 8, &sB[lrow][lcol + 8]);
    async_wait0();                              // s_wait_asynccnt 0
    __syncthreads();

    FragU fa[2], fb[4];
    // A fragment (16x32): lane lt = row m; chunks K = hl*8..+7 and 16+hl*8..+7
    for (int mt = 0; mt < 2; mt++) {
      int r = wm * 32 + mt * 16 + lt;
      fa[mt].q[0] = *(const v4u*)&sA[r][hl * 8];
      fa[mt].q[1] = *(const v4u*)&sA[r][16 + hl * 8];
    }
    // B fragment (32x16): lane lt = col n; chunks K = hl*8..+7 and 16+hl*8..+7
    for (int nt = 0; nt < 4; nt++) {
      int r = wn * 64 + nt * 16 + lt;
      fb[nt].q[0] = *(const v4u*)&sB[r][hl * 8];
      fb[nt].q[1] = *(const v4u*)&sB[r][16 + hl * 8];
    }
    for (int mt = 0; mt < 2; mt++)
      for (int nt = 0; nt < 4; nt++)
        acc[mt][nt] = wmma_bf16(fa[mt].v, fb[nt].v, acc[mt][nt]);
  }

  // Epilogue. C layout: lane lt = column n, hl selects m=j+8*hl.
  for (int nt = 0; nt < 4; nt++) {
    const int col = n0 + wn * 64 + nt * 16 + lt;
    const float bv1 = bias1 ? bias1[col] : 0.f;
    const float bv2 = bias2 ? bias2[col] : 0.f;
    for (int mt = 0; mt < 2; mt++) {
      for (int j = 0; j < 8; j++) {
        const int row = m0 + wm * 32 + mt * 16 + j + 8 * hl;
        const size_t idx = cofs + (size_t)row * ldc + col;
        const float v = acc[mt][nt][j] * scale + bv1;
        if (outf) outf[idx] = v;
        else      out1[idx] = (__bf16)v;
        if (out2) out2[idx] = (__bf16)(v + bv2);
      }
    }
  }
}

// ---------------------------------------------------------------------------
// content bias: cb[b,h,t] = scale * sum_d r_w_bias[h,d] * k[b,t,h,d]
// ---------------------------------------------------------------------------
__global__ void content_bias_kernel(const __bf16* __restrict__ kb,
                                    const float* __restrict__ rwb,
                                    float* __restrict__ cbuf, float scale) {
  int i = blockIdx.x * 256 + threadIdx.x;       // i = b*32768 + h*2048 + t
  int tpos = i & 2047, h = (i >> 11) & 15, b = i >> 15;
  const __bf16* krow = kb + ((size_t)(b * 2048 + tpos)) * 1024 + h * 64;
  float s = 0.f;
  for (int d = 0; d < 64; d++) s += (float)krow[d] * rwb[h * 64 + d];
  cbuf[i] = s * scale;
}

// ---------------------------------------------------------------------------
// Flash attention with Transformer-XL relative shift.
// Rel-shift closed form (Q==K==S):
//   PB[q,k] = E[q, k-q+S-1]  (k<=q);  0 (k==q+1);  E[q+1, k-q-2]  (k>q+1)
// grid = (S/128, B*H); block = 256 (8 waves); wave owns 16 q rows; K chunk 64.
// ---------------------------------------------------------------------------
__global__ __launch_bounds__(256) void relattn_kernel(
    const __bf16* __restrict__ qb, const __bf16* __restrict__ kb,
    const __bf16* __restrict__ vb, const __bf16* __restrict__ Ebuf,
    const float* __restrict__ cbuf, __bf16* __restrict__ ao, float scale)
{
  constexpr int S = 2048, HD = 64, NH = 16;
  __shared__ __bf16 sK[64][72];      // [pos][d]    72*2 = 144B rows (16B aligned)
  __shared__ __bf16 sVt[64][72];     // [d][pos]
  __shared__ __bf16 sP[8][16][72];   // per-wave P tile [qrow][pos]

  const int t = threadIdx.x;
  const int w = t >> 5, lane = t & 31, lt = lane & 15, hl = lane >> 4;
  const int b = blockIdx.y >> 4, h = blockIdx.y & 15;
  const int q0 = blockIdx.x * 128 + w * 16;

  // Q fragments (A layout), loaded once: rows q0..q0+15, d split in two K-steps
  FragU fq[2];
  const __bf16* qrow = qb + ((size_t)(b * S + q0 + lt)) * 1024 + h * HD;
  for (int s = 0; s < 2; s++) {
    fq[s].q[0] = *(const v4u*)(qrow + s * 32 + hl * 8);
    fq[s].q[1] = *(const v4u*)(qrow + s * 32 + 16 + hl * 8);
  }

  const v8f vzero = {0.f,0.f,0.f,0.f,0.f,0.f,0.f,0.f};
  float mrow[8], lrowv[8];
  v8f oacc[4];
  for (int j = 0; j < 8; j++) { mrow[j] = -1e30f; lrowv[j] = 0.f; }
  for (int d = 0; d < 4; d++) oacc[d] = vzero;

  const size_t kvbase = (size_t)(b * S) * 1024 + h * HD;
  const size_t ebase  = ((size_t)(b * NH + h)) * S * S;
  const float* cbh    = cbuf + (size_t)(b * NH + h) * S;

  for (int c0 = 0; c0 < S; c0 += 64) {
    // ---- stage K (async DMA to LDS) and V (transposed, via VGPRs) ----
    {
      int pos = t >> 2, dcol = (t & 3) * 16;
      const __bf16* kp = kb + kvbase + (size_t)(c0 + pos) * 1024 + dcol;
      async16(kp,     &sK[pos][dcol]);
      async16(kp + 8, &sK[pos][dcol + 8]);
      const __bf16* vp = vb + kvbase + (size_t)(c0 + pos) * 1024 + dcol;
      union { v4u u[2]; __bf16 e[16]; } tv;
      tv.u[0] = *(const v4u*)vp; tv.u[1] = *(const v4u*)(vp + 8);
      for (int i = 0; i < 16; i++) sVt[dcol + i][pos] = tv.e[i];
      async_wait0();
    }
    __syncthreads();

    // ---- AC = q . k^T via WMMA; add scale, content bias, shifted pos bias ----
    float sval[4][8];
    for (int nt = 0; nt < 4; nt++) {
      v8f a = vzero;
      for (int s = 0; s < 2; s++) {
        FragU fb;
        int prow = nt * 16 + lt;                 // B: lane = column (k position)
        fb.q[0] = *(const v4u*)&sK[prow][s * 32 + hl * 8];
        fb.q[1] = *(const v4u*)&sK[prow][s * 32 + 16 + hl * 8];
        a = wmma_bf16(fq[s].v, fb.v, a);
      }
      const int kg = c0 + nt * 16 + lt;
      const float cbv = cbh[kg];
      for (int j = 0; j < 8; j++) {
        const int qg = q0 + j + 8 * hl;
        float pb = 0.f;
        if (kg != qg + 1) {
          int rq, tt;
          if (kg <= qg) { rq = qg;     tt = kg - qg + (S - 1); }
          else          { rq = qg + 1; tt = kg - qg - 2;       }
          pb = (float)Ebuf[ebase + (size_t)rq * S + tt];   // E pre-scaled
        }
        sval[nt][j] = a[j] * scale + cbv + pb;
      }
    }

    // ---- online softmax (row reductions across 16 lanes of each half) ----
    float pcorr[8];
    for (int j = 0; j < 8; j++) {
      float mx = fmaxf(fmaxf(sval[0][j], sval[1][j]), fmaxf(sval[2][j], sval[3][j]));
      for (int off = 1; off < 16; off <<= 1) mx = fmaxf(mx, __shfl_xor(mx, off, 32));
      const float mnew = fmaxf(mrow[j], mx);
      const float corr = __expf(mrow[j] - mnew);
      mrow[j] = mnew;
      float rs = 0.f;
      for (int nt = 0; nt < 4; nt++) {
        float p = __expf(sval[nt][j] - mnew);
        sval[nt][j] = p;
        rs += p;
      }
      for (int off = 1; off < 16; off <<= 1) rs += __shfl_xor(rs, off, 32);
      lrowv[j] = lrowv[j] * corr + rs;
      pcorr[j] = corr;
    }
    for (int d = 0; d < 4; d++)
      for (int j = 0; j < 8; j++) oacc[d][j] *= pcorr[j];

    // ---- C-layout -> A-layout for P via wave-private LDS ----
    for (int nt = 0; nt < 4; nt++)
      for (int j = 0; j < 8; j++)
        sP[w][j + 8 * hl][nt * 16 + lt] = (__bf16)sval[nt][j];
    // same-wave DS ops retire in order; compiler inserts s_wait_dscnt

    // ---- PV: O(16x64) += P(16x64) @ V(64x64) ----
    for (int s2 = 0; s2 < 2; s2++) {
      FragU fp;
      fp.q[0] = *(const v4u*)&sP[w][lt][s2 * 32 + hl * 8];
      fp.q[1] = *(const v4u*)&sP[w][lt][s2 * 32 + 16 + hl * 8];
      for (int dt = 0; dt < 4; dt++) {
        FragU fv;
        fv.q[0] = *(const v4u*)&sVt[dt * 16 + lt][s2 * 32 + hl * 8];
        fv.q[1] = *(const v4u*)&sVt[dt * 16 + lt][s2 * 32 + 16 + hl * 8];
        oacc[dt] = wmma_bf16(fp.v, fv.v, oacc[dt]);
      }
    }
    __syncthreads();
  }

  // ---- normalize and store attn output [b, q, h*64+d] bf16 ----
  float inv[8];
  for (int j = 0; j < 8; j++) inv[j] = 1.f / lrowv[j];
  for (int dt = 0; dt < 4; dt++) {
    int d = dt * 16 + lt;
    for (int j = 0; j < 8; j++) {
      int row = q0 + j + 8 * hl;
      ao[((size_t)(b * S + row)) * 1024 + h * HD + d] = (__bf16)(oacc[dt][j] * inv[j]);
    }
  }
}

// ---------------------------------------------------------------------------
// Host launcher
// ---------------------------------------------------------------------------
extern "C" void kernel_launch(void* const* d_in, const int* in_sizes, int n_in,
                              void* d_out, int out_size, void* d_ws, size_t ws_size,
                              hipStream_t stream) {
  const float* inputs_kv = (const float*)d_in[0];
  const float* inputs_q  = (const float*)d_in[1];
  const float* pos_embed = (const float*)d_in[2];
  const float* Wq_w = (const float*)d_in[3];
  const float* Wq_b = (const float*)d_in[4];
  const float* Wk_w = (const float*)d_in[5];
  const float* Wk_b = (const float*)d_in[6];
  const float* Wv_w = (const float*)d_in[7];
  const float* Wv_b = (const float*)d_in[8];
  const float* Wr_w = (const float*)d_in[9];
  const float* r_w_bias = (const float*)d_in[10];
  const float* r_r_bias = (const float*)d_in[11];  // flat [1024] = per-column bias
  const float* Wo_w = (const float*)d_in[12];
  const float* Wo_b = (const float*)d_in[13];
  float* out = (float*)d_out;

  constexpr int B_ = 2, S = 2048, D = 1024, H_ = 16, HD_ = 64;
  const float scale = 0.125f;  // 64^-0.5

  char* p = (char*)d_ws;
  auto alloc = [&](size_t bytes) -> char* {
    char* r = p; p += (bytes + 255) & ~(size_t)255; return r;
  };
  __bf16* wq_t = (__bf16*)alloc((size_t)D * D * 2);
  __bf16* wk_t = (__bf16*)alloc((size_t)D * D * 2);
  __bf16* wv_t = (__bf16*)alloc((size_t)D * D * 2);
  __bf16* wr_t = (__bf16*)alloc((size_t)D * D * 2);
  __bf16* wo_t = (__bf16*)alloc((size_t)D * D * 2);
  __bf16* xq   = (__bf16*)alloc((size_t)B_ * S * D * 2);
  __bf16* xkv  = (__bf16*)alloc((size_t)B_ * S * D * 2);
  __bf16* posb = (__bf16*)alloc((size_t)S * D * 2);
  __bf16* qbuf = (__bf16*)alloc((size_t)B_ * S * D * 2);
  __bf16* qhat = (__bf16*)alloc((size_t)B_ * S * D * 2);
  __bf16* kbuf = (__bf16*)alloc((size_t)B_ * S * D * 2);
  __bf16* vbuf = (__bf16*)alloc((size_t)B_ * S * D * 2);
  __bf16* rbuf = (__bf16*)alloc((size_t)S * D * 2);
  float*  cbuf = (float*) alloc((size_t)B_ * H_ * S * 4);
  __bf16* ao   = (__bf16*)alloc((size_t)B_ * S * D * 2);
  __bf16* Ebuf = (__bf16*)alloc((size_t)B_ * H_ * S * S * 2);  // 256 MB
  if ((size_t)(p - (char*)d_ws) > ws_size) return;  // workspace too small

  // 1) conversions
  to_bf16_vec<<<(B_ * S * D) / 1024, 256, 0, stream>>>(inputs_q, xq, B_ * S * D);
  to_bf16_vec<<<(B_ * S * D) / 1024, 256, 0, stream>>>(inputs_kv, xkv, B_ * S * D);
  to_bf16_vec<<<(S * D) / 1024, 256, 0, stream>>>(pos_embed, posb, S * D);
  transpose_to_bf16<<<(D * D) / 256, 256, 0, stream>>>(Wq_w, wq_t, D, D);
  transpose_to_bf16<<<(D * D) / 256, 256, 0, stream>>>(Wk_w, wk_t, D, D);
  transpose_to_bf16<<<(D * D) / 256, 256, 0, stream>>>(Wv_w, wv_t, D, D);
  transpose_to_bf16<<<(D * D) / 256, 256, 0, stream>>>(Wr_w, wr_t, D, D);
  transpose_to_bf16<<<(D * D) / 256, 256, 0, stream>>>(Wo_w, wo_t, D, D);

  // 2) projections: q (+ qhat = q + r_r_bias), k, v, r
  dim3 gp(D / 128, (B_ * S) / 128, 1);
  gemm_bt_bf16<<<gp, 256, 0, stream>>>(xq, D, 0, 0, wq_t, D, 0, 0,
                                       Wq_b, r_r_bias, qbuf, qhat, nullptr,
                                       D, 0, 0, D, 1, 1.f);
  gemm_bt_bf16<<<gp, 256, 0, stream>>>(xkv, D, 0, 0, wk_t, D, 0, 0,
                                       Wk_b, nullptr, kbuf, nullptr, nullptr,
                                       D, 0, 0, D, 1, 1.f);
  gemm_bt_bf16<<<gp, 256, 0, stream>>>(xkv, D, 0, 0, wv_t, D, 0, 0,
                                       Wv_b, nullptr, vbuf, nullptr, nullptr,
                                       D, 0, 0, D, 1, 1.f);
  dim3 gr(D / 128, S / 128, 1);
  gemm_bt_bf16<<<gr, 256, 0, stream>>>(posb, D, 0, 0, wr_t, D, 0, 0,
                                       nullptr, nullptr, rbuf, nullptr, nullptr,
                                       D, 0, 0, D, 1, 1.f);

  // 3) content bias cb[b,h,t]
  content_bias_kernel<<<(B_ * H_ * S) / 256, 256, 0, stream>>>(kbuf, r_w_bias, cbuf, scale);

  // 4) E[b,h,q,t] = scale * qhat . r^T  (batched GEMM over b,h via strides)
  dim3 ge(S / 128, S / 128, B_ * H_);
  gemm_bt_bf16<<<ge, 256, 0, stream>>>(
      qhat, D, (long long)S * D, HD_,      // A: batch b stride S*D, head stride 64
      rbuf, D, 0, HD_,                     // Bt: r[t][h*64+d], head stride 64
      nullptr, nullptr, Ebuf, nullptr, nullptr,
      S, (long long)H_ * S * S, (long long)S * S,
      HD_, H_, scale);

  // 5) flash attention with rel-shift gather
  dim3 ga(S / 128, B_ * H_);
  relattn_kernel<<<ga, 256, 0, stream>>>(qbuf, kbuf, vbuf, Ebuf, cbuf, ao, scale);

  // 6) output projection -> fp32 d_out
  gemm_bt_bf16<<<gp, 256, 0, stream>>>(ao, D, 0, 0, wo_t, D, 0, 0,
                                       Wo_b, nullptr, nullptr, nullptr, out,
                                       D, 0, 0, D, 1, 1.f);
}
